// Wide_BasicBlock_Q_13477607375128
// MI455X (gfx1250) — 8-bit matrix operand order repaired and checked by reading, compiled, not run on hardware
//
#include <hip/hip_runtime.h>
#include <stdint.h>

typedef int      v8i  __attribute__((ext_vector_type(8)));
typedef int      v4i  __attribute__((ext_vector_type(4)));
typedef unsigned v4u  __attribute__((ext_vector_type(4)));

#define BN_EPS 1e-5f

__device__ __forceinline__ unsigned umax32(unsigned a, unsigned b) { return a > b ? a : b; }

#if __has_builtin(__builtin_amdgcn_tensor_load_to_lds)
#define ATHENA_USE_TDM 1
__device__ __forceinline__ void tdm_load_1d(unsigned lds_off, const void* gptr, unsigned nbytes) {
  unsigned long long ga = (unsigned long long)(uintptr_t)gptr;
  v4u g0;
  g0[0] = 1u;
  g0[1] = lds_off;
  g0[2] = (unsigned)(ga & 0xFFFFFFFFu);
  g0[3] = (unsigned)((ga >> 32) & 0x01FFFFFFu) | (2u << 30);
  unsigned td0 = 0x40000000u;
  v8i g1;
  g1[0] = 0;
  g1[1] = (int)((td0 & 0xFFFFu) << 16);
  g1[2] = (int)(((td0 >> 16) & 0xFFFFu) | (1u << 16));
  g1[3] = (int)((nbytes & 0xFFFFu) << 16);
  g1[4] = 0;
  g1[5] = (int)td0;
  g1[6] = 0;
  g1[7] = 0;
  v4i z4 = {};
#if __clang_major__ >= 23
  v8i z8 = {};
  __builtin_amdgcn_tensor_load_to_lds(g0, g1, z4, z4, z8, 0);
#else
  __builtin_amdgcn_tensor_load_to_lds(g0, g1, z4, z4, 0);
#endif
}
#else
#define ATHENA_USE_TDM 0
#endif

__global__ void prep_kernel(const float* g1, const float* b1, const float* m1, const float* v1,
                            const float* g2, const float* b2, const float* m2, const float* v2,
                            const float* gs, const float* bs, const float* ms, const float* vs,
                            float* s1, float* sh1, float* s2, float* sh2,
                            float* scs, float* shs, unsigned* maxslots) {
  int t = threadIdx.x;
  if (t < 64) {
    float s = g1[t] * rsqrtf(v1[t] + BN_EPS);
    s1[t] = s;  sh1[t] = b1[t] - m1[t] * s;
    float ss = gs[t] * rsqrtf(vs[t] + BN_EPS);
    scs[t] = ss; shs[t] = bs[t] - ms[t] * ss;
  }
  if (t < 128) {
    float s = g2[t] * rsqrtf(v2[t] + BN_EPS);
    s2[t] = s;  sh2[t] = b2[t] - m2[t] * s;
  }
  if (t < 2) maxslots[t] = 0u;
}

__global__ void absmax_tanh_kernel(const float* __restrict__ w, int n, unsigned* slot) {
  __shared__ unsigned red[256];
  unsigned best = 0u;
  for (int i = blockIdx.x * blockDim.x + threadIdx.x; i < n; i += gridDim.x * blockDim.x)
    best = umax32(best, __float_as_uint(fabsf(tanhf(w[i]))));
  red[threadIdx.x] = best;
  __syncthreads();
  for (int s = 128; s > 0; s >>= 1) {
    if (threadIdx.x < (unsigned)s) red[threadIdx.x] = umax32(red[threadIdx.x], red[threadIdx.x + s]);
    __syncthreads();
  }
  if (threadIdx.x == 0) atomicMax(slot, red[0]);
}

__global__ void quantw_kernel(const float* __restrict__ w, int8_t* __restrict__ wq,
                              int Cout, int Cin, const unsigned* slot) {
  int total = Cout * Cin * 9;
  float invden = 1.f / (2.f * __uint_as_float(*slot));
  for (int i = blockIdx.x * blockDim.x + threadIdx.x; i < total; i += gridDim.x * blockDim.x) {
    int t   = i / (Cout * Cin);
    int rem = i % (Cout * Cin);
    int co = rem / Cin, ci = rem % Cin;
    int kh = t / 3, kw = t % 3;
    float xv = w[((co * Cin + ci) * 3 + kh) * 3 + kw];
    float q = rintf((tanhf(xv) * invden + 0.5f) * 15.f);
    q = fminf(fmaxf(q, 0.f), 15.f);
    wq[i] = (int8_t)(2 * (int)q - 15);
  }
}

__global__ void act1_pad_kernel(const float* __restrict__ x, uint8_t* __restrict__ a1p,
                                const float* __restrict__ s1, const float* __restrict__ sh1) {
  int i = blockIdx.x * blockDim.x + threadIdx.x;
  int c  = i & 63;
  int sp = i >> 6;
  int px = sp % 34;
  int t  = sp / 34;
  int py = t % 34;
  int n  = t / 34;
  int ix = px - 1, iy = py - 1;
  uint8_t q = 0;
  if ((unsigned)ix < 32u && (unsigned)iy < 32u) {
    float v = x[((n * 64 + c) * 32 + iy) * 32 + ix];
    v = v * s1[c] + sh1[c];
    v = fminf(fmaxf(v, 0.f), 1.f);
    q = (uint8_t)(int)rintf(v * 15.f);
  }
  a1p[i] = q;
}

__global__ void halo2_kernel(uint8_t* __restrict__ a2p) {
  int i = blockIdx.x * blockDim.x + threadIdx.x;
  int c = i & 127;
  int r = i >> 7;
  int n = r / 68, p = r % 68;
  int y, xx;
  if (p < 18)      { y = 0;          xx = p;      }
  else if (p < 36) { y = 17;         xx = p - 18; }
  else if (p < 52) { y = p - 36 + 1; xx = 0;      }
  else             { y = p - 52 + 1; xx = 17;     }
  a2p[((n * 18 + y) * 18 + xx) * 128 + c] = 0;
}

__global__ __launch_bounds__(256) void conv1_kernel(const uint8_t* __restrict__ a1p,
                                                    const int8_t* __restrict__ wq1,
                                                    uint8_t* __restrict__ a2p,
                                                    const float* __restrict__ s2,
                                                    const float* __restrict__ sh2) {
  __shared__ uint8_t lds[3 * 2176];
  int n = blockIdx.x >> 4, oy = blockIdx.x & 15;
  int tid = threadIdx.x;
  const uint8_t* gsrc = a1p + (size_t)((n * 34 + 2 * oy) * 34) * 64;
  int wv = tid >> 5;
  __builtin_prefetch(wq1 + ((wv * 16) << 6), 0, 3);

#if ATHENA_USE_TDM
  if (tid < 32) {
    tdm_load_1d((unsigned)(uintptr_t)lds, gsrc, 3 * 2176);
    __builtin_amdgcn_s_wait_tensorcnt((short)0);
  }
#else
  for (int d = tid; d < 1632; d += 256)
    *(unsigned*)(lds + 4 * d) = *(const unsigned*)(gsrc + 4 * d);
#endif
  __syncthreads();

  int lane = tid & 31;
  int l15 = lane & 15, g = lane >> 4;
  int kA[8], kB[8];
#pragma unroll
  for (int v = 0; v < 8; ++v) {
    kA[v] = (v & 1) * 4 + ((v >> 1) & 1) * 16 + g * 8 + (v >> 2) * 32;
    kB[v] = (v & 1) * 4 + ((v >> 1) & 1) * 16 + g * 8 + (v >> 2) * 32;
  }

  v8i acc0 = {}, acc1 = {};
#pragma unroll
  for (int kh = 0; kh < 3; ++kh) {
#pragma unroll
    for (int kw = 0; kw < 3; ++kw) {
      int t = kh * 3 + kw;
      int p = 2 * l15 + kw;
      const uint8_t* abase = lds + kh * 2176 + p * 64;
      const int8_t* bbase = wq1 + ((t * 128 + wv * 16 + l15) << 6);
      v8i a, b;
#pragma unroll
      for (int v = 0; v < 8; ++v) {
        a[v] = *(const int*)(abase + kA[v]);
        b[v] = *(const int*)(bbase + kB[v]);
      }
      if (t & 1)
        acc1 = __builtin_amdgcn_wmma_i32_16x16x64_iu8(false, a, true, b, acc1, false, false);
      else
        acc0 = __builtin_amdgcn_wmma_i32_16x16x64_iu8(false, a, true, b, acc0, false, false);
    }
  }
  v8i acc = acc0 + acc1;

  int cout = wv * 16 + l15;
  float sc = s2[cout] * (1.f / 225.f);
  float sh = sh2[cout];
#pragma unroll
  for (int r = 0; r < 8; ++r) {
    int oxp = r + 8 * g;
    float v = (float)acc[r] * sc + sh;
    v = fminf(fmaxf(v, 0.f), 1.f);
    a2p[(((n * 18 + oy + 1) * 18 + oxp + 1) << 7) + cout] = (uint8_t)(int)rintf(v * 15.f);
  }
}

__global__ __launch_bounds__(256) void shortcut_kernel(const float* __restrict__ x,
                                                       const float* __restrict__ wsw,
                                                       const float* __restrict__ scs,
                                                       const float* __restrict__ shs,
                                                       float* __restrict__ out) {
  __shared__ float lds[64 * 16];
  int n = blockIdx.x >> 4, oy = blockIdx.x & 15;
  int tid = threadIdx.x;
  for (int d = tid; d < 1024; d += 256) {
    int k = d >> 4, ox = d & 15;
    float v = x[((n * 64 + k) * 32 + 2 * oy) * 32 + 2 * ox];
    lds[d] = v * scs[k] + shs[k];
  }
  __syncthreads();
  for (int i = tid; i < 2048; i += 256) {
    int c = i >> 4, ox = i & 15;
    const float* wrow = wsw + c * 64;
    float acc = 0.f;
#pragma unroll 16
    for (int k = 0; k < 64; ++k) acc += lds[k * 16 + ox] * wrow[k];
    out[((n * 128 + c) * 16 + oy) * 16 + ox] = acc;
  }
}

__global__ __launch_bounds__(256) void conv2_kernel(const uint8_t* __restrict__ a2p,
                                                    const int8_t* __restrict__ wq2,
                                                    float* __restrict__ out) {
  __shared__ uint8_t lds[3 * 2304];
  int n = blockIdx.x >> 4, oy = blockIdx.x & 15;
  int tid = threadIdx.x;
  const uint8_t* gsrc = a2p + (size_t)((n * 18 + oy) * 18) * 128;
  int wv = tid >> 5;
  __builtin_prefetch(wq2 + ((wv * 16) << 7), 0, 3);

#if ATHENA_USE_TDM
  if (tid < 32) {
    tdm_load_1d((unsigned)(uintptr_t)lds, gsrc, 3 * 2304);
    __builtin_amdgcn_s_wait_tensorcnt((short)0);
  }
#else
  for (int d = tid; d < 1728; d += 256)
    *(unsigned*)(lds + 4 * d) = *(const unsigned*)(gsrc + 4 * d);
#endif
  __syncthreads();

  int lane = tid & 31;
  int l15 = lane & 15, g = lane >> 4;
  int kA[8], kB[8];
#pragma unroll
  for (int v = 0; v < 8; ++v) {
    kA[v] = (v & 1) * 4 + ((v >> 1) & 1) * 16 + g * 8 + (v >> 2) * 32;
    kB[v] = (v & 1) * 4 + ((v >> 1) & 1) * 16 + g * 8 + (v >> 2) * 32;
  }

  v8i acc0 = {}, acc1 = {};
#pragma unroll
  for (int kh = 0; kh < 3; ++kh) {
#pragma unroll
    for (int kw = 0; kw < 3; ++kw) {
      int t = kh * 3 + kw;
      int p = l15 + kw;
#pragma unroll
      for (int ch = 0; ch < 2; ++ch) {
        const uint8_t* abase = lds + kh * 2304 + p * 128 + ch * 64;
        const int8_t* bbase = wq2 + ((t * 128 + wv * 16 + l15) << 7) + (ch << 6);
        v8i a, b;
#pragma unroll
        for (int v = 0; v < 8; ++v) {
          a[v] = *(const int*)(abase + kA[v]);
          b[v] = *(const int*)(bbase + kB[v]);
        }
        if ((t * 2 + ch) & 1)
          acc1 = __builtin_amdgcn_wmma_i32_16x16x64_iu8(false, a, true, b, acc1, false, false);
        else
          acc0 = __builtin_amdgcn_wmma_i32_16x16x64_iu8(false, a, true, b, acc0, false, false);
      }
    }
  }
  v8i acc = acc0 + acc1;

  int cout = wv * 16 + l15;
#pragma unroll
  for (int r = 0; r < 8; ++r) {
    int oxp = r + 8 * g;
    int idx = ((n * 128 + cout) * 16 + oy) * 16 + oxp;
    out[idx] += (float)acc[r] * (1.f / 225.f);
  }
}

extern "C" void kernel_launch(void* const* d_in, const int* in_sizes, int n_in,
                              void* d_out, int out_size, void* d_ws, size_t ws_size,
                              hipStream_t stream) {
  const float* x   = (const float*)d_in[0];
  const float* w1  = (const float*)d_in[1];
  const float* w2  = (const float*)d_in[2];
  const float* wsw = (const float*)d_in[3];
  const float* g1  = (const float*)d_in[4];
  const float* b1  = (const float*)d_in[5];
  const float* m1  = (const float*)d_in[6];
  const float* v1  = (const float*)d_in[7];
  const float* g2  = (const float*)d_in[8];
  const float* b2  = (const float*)d_in[9];
  const float* m2  = (const float*)d_in[10];
  const float* v2  = (const float*)d_in[11];
  const float* gs  = (const float*)d_in[12];
  const float* bs  = (const float*)d_in[13];
  const float* ms  = (const float*)d_in[14];
  const float* vs  = (const float*)d_in[15];
  float* out = (float*)d_out;

  uint8_t* base = (uint8_t*)d_ws;
  uint8_t* a1p = base;
  uint8_t* a2p = base + 18939904;
  int8_t*  wq1 = (int8_t*)(base + 29556736);
  int8_t*  wq2 = (int8_t*)(base + 29630464);
  float*   s1  = (float*)(base + 29777920);
  float*   sh1 = s1 + 64;
  float*   s2  = sh1 + 64;
  float*   sh2 = s2 + 128;
  float*   scs = sh2 + 128;
  float*   shs = scs + 64;
  unsigned* maxslots = (unsigned*)(shs + 64);

  prep_kernel<<<1, 128, 0, stream>>>(g1, b1, m1, v1, g2, b2, m2, v2, gs, bs, ms, vs,
                                     s1, sh1, s2, sh2, scs, shs, maxslots);
  absmax_tanh_kernel<<<64, 256, 0, stream>>>(w1, 128 * 64 * 9,  maxslots + 0);
  absmax_tanh_kernel<<<64, 256, 0, stream>>>(w2, 128 * 128 * 9, maxslots + 1);
  quantw_kernel<<<128, 256, 0, stream>>>(w1, wq1, 128, 64,  maxslots + 0);
  quantw_kernel<<<256, 256, 0, stream>>>(w2, wq2, 128, 128, maxslots + 1);
  act1_pad_kernel<<<73984, 256, 0, stream>>>(x, a1p, s1, sh1);
  halo2_kernel<<<8704, 256, 0, stream>>>(a2p);
  conv1_kernel<<<4096, 256, 0, stream>>>(a1p, wq1, a2p, s2, sh2);
  shortcut_kernel<<<4096, 256, 0, stream>>>(x, wsw, scs, shs, out);
  conv2_kernel<<<4096, 256, 0, stream>>>(a2p, wq2, out);
}
